// _TRMMultiHeadAttention_61332132987366
// MI455X (gfx1250) — matrix operand order repaired and checked by reading, not compiled, not run on hardware
//
#include <hip/hip_runtime.h>
#include <cstdint>

// ---------------------------------------------------------------------------
// Problem constants (from reference): B=4, S=2048, E=1024, H=16, HD=64
// ---------------------------------------------------------------------------
#define BB   4
#define SS   2048
#define EE   1024
#define HH   16
#define HDD  64
#define NE3  (3 * EE)          // 3072
#define MM   (BB * SS)         // 8192 rows

typedef __attribute__((ext_vector_type(16))) _Float16 v16h;
typedef __attribute__((ext_vector_type(8)))  _Float16 v8h;
typedef __attribute__((ext_vector_type(4)))  _Float16 v4h;
typedef __attribute__((ext_vector_type(8)))  float    v8f;
typedef __attribute__((ext_vector_type(4)))  int      v4i;

#define WMMA_F16(a, b, c) \
    __builtin_amdgcn_wmma_f32_16x16x32_f16(false, (a), false, (b), (short)0, (c), false, false)

// LDS k-stride (halves). 40 halves = 80B: b128 fragment loads are 16B-aligned
// and bank = (20*col) mod 64 is conflict-free across a 16-lane half-wave.
#define BST    40
#define TILE_H (64 * BST)      // halves per staging buffer (64 cols x 32 k + pad)

// CDNA5 async global->LDS staging (ASYNCcnt), with portable fallback.
#if defined(__HIP_DEVICE_COMPILE__) && \
    __has_builtin(__builtin_amdgcn_global_load_async_to_lds_b128) && \
    __has_builtin(__builtin_amdgcn_s_wait_asynccnt)
#define HAVE_ASYNC 1
#else
#define HAVE_ASYNC 0
#endif

#if HAVE_ASYNC
// Builtin signature (from hipcc diagnostic): (v4i AS1*, v4i AS3*, imm, imm)
typedef __attribute__((address_space(1))) v4i gas_v4i;
typedef __attribute__((address_space(3))) v4i las_v4i;

__device__ __forceinline__ void stage_async(const _Float16* g, _Float16* l) {
    __builtin_amdgcn_global_load_async_to_lds_b128(
        (gas_v4i*)(uintptr_t)g,
        (las_v4i*)(uint32_t)(uintptr_t)l,
        0, 0);
}
#endif

// Concatenate two v8h loads into one 16-half WMMA operand.
__device__ __forceinline__ v16h load_frag_pair(const _Float16* p0, const _Float16* p1) {
    v8h lo = *(const v8h*)p0;
    v8h hi = *(const v8h*)p1;
    v16h r;
#pragma unroll
    for (int j = 0; j < 8; ++j) { r[j] = lo[j]; r[8 + j] = hi[j]; }
    return r;
}

// ---------------------------------------------------------------------------
// fp32 -> f16 conversion (vectorized x4) for the activation stream
// ---------------------------------------------------------------------------
__global__ __launch_bounds__(256) void cvt_kernel(const float4* __restrict__ in,
                                                  v4h* __restrict__ out, int n4) {
    int i = blockIdx.x * 256 + threadIdx.x;
    int stride = gridDim.x * 256;
    for (; i < n4; i += stride) {
        float4 a = in[i];
        v4h o;
        o[0] = (_Float16)a.x; o[1] = (_Float16)a.y;
        o[2] = (_Float16)a.z; o[3] = (_Float16)a.w;
        out[i] = o;
    }
}

// ---------------------------------------------------------------------------
// fp32 [K][N] -> f16 transposed [N][K] (one-time, so GEMM B staging is a
// contiguous global->LDS copy; enables async-to-LDS). LDS-tiled 64x64.
// ---------------------------------------------------------------------------
__global__ __launch_bounds__(256) void cvtT_kernel(const float* __restrict__ in,
                                                   _Float16* __restrict__ out,
                                                   int K, int N) {
    __shared__ _Float16 tile[64][72];   // +8 pad: declash banks on column reads
    const int n0 = blockIdx.x * 64;
    const int k0 = blockIdx.y * 64;
    const int tr = threadIdx.x >> 4;        // 0..15
    const int tc = (threadIdx.x & 15) * 4;  // 0..60
#pragma unroll
    for (int i = 0; i < 4; ++i) {
        const int k = tr + i * 16;
        const float4 a = *(const float4*)(in + (long)(k0 + k) * N + n0 + tc);
        v4h o;
        o[0] = (_Float16)a.x; o[1] = (_Float16)a.y;
        o[2] = (_Float16)a.z; o[3] = (_Float16)a.w;
        *(v4h*)&tile[k][tc] = o;
    }
    __syncthreads();
#pragma unroll
    for (int i = 0; i < 4; ++i) {
        const int n = tr + i * 16;
        v4h o;
#pragma unroll
        for (int j = 0; j < 4; ++j) o[j] = tile[tc + j][n];
        *(v4h*)(out + (long)(n0 + n) * K + k0 + tc) = o;
    }
}

// ---------------------------------------------------------------------------
// Shared GEMM mainloop: C[128x64] = A[128xK] * W[N x K]^T tile at (Mbase,Nbase).
// WT is [N][K] row-major. Double-buffered LDS staging (async-to-LDS when
// available), one barrier per K-step, batched B-fragments, back-to-back WMMAs.
// ---------------------------------------------------------------------------
__device__ __forceinline__ void gemm_mainloop(const _Float16* __restrict__ A,
                                              const _Float16* __restrict__ WT,
                                              int K, int Mbase, int Nbase,
                                              _Float16* lds, v8f acc[4]) {
    const int t    = threadIdx.x;
    const int wave = t >> 5;
    const int lane = t & 31;
    const int lm   = lane & 15;
    const int lg   = lane >> 4;
    const long rowA = Mbase + wave * 16 + lm;

    // Staging assignment: thread copies 8 halves (16B) of WT row (Nbase+sn),
    // k-segment sk, into lds[sn*BST + sk].
    const int sn = t >> 2;          // 0..63
    const int sk = (t & 3) * 8;     // 0,8,16,24
    const _Float16* gstage = WT + (long)(Nbase + sn) * K + sk;
    const int lofs = sn * BST + sk;

    // Prologue: stage tile 0 into buffer 0.
#if HAVE_ASYNC
    stage_async(gstage, lds + lofs);
    __builtin_amdgcn_s_wait_asynccnt(0);
#else
    *(v8h*)(lds + lofs) = *(const v8h*)gstage;
#endif
    __syncthreads();

    for (int k0 = 0; k0 < K; k0 += 32) {
        _Float16* cur = lds + ((k0 >> 5) & 1) * TILE_H;
        _Float16* nxt = lds + ((((k0 >> 5) & 1)) ^ 1) * TILE_H;
        const bool more = (k0 + 32) < K;

        // Kick off staging of the next tile (hidden behind this tile's WMMAs).
#if HAVE_ASYNC
        if (more) stage_async(gstage + k0 + 32, nxt + lofs);
#else
        v8h wreg;
        if (more) wreg = *(const v8h*)(gstage + k0 + 32);
#endif
        if (k0 + 64 < K) __builtin_prefetch(gstage + k0 + 64, 0, 1);

        // A fragment: 16x32, interleaved K layout (ISA 7.12.2).
        const _Float16* ap = A + rowA * K + k0 + lg * 8;
        const v16h af = load_frag_pair(ap, ap + 16);

        // All four B fragments first (one dscnt wait covers 8 ds_load_b128),
        // then four back-to-back WMMAs (distinct D/C, shared A: no hazards).
        v16h bf[4];
#pragma unroll
        for (int nt = 0; nt < 4; ++nt) {
            const _Float16* bp = cur + (nt * 16 + lm) * BST + lg * 8;
            bf[nt] = load_frag_pair(bp, bp + 16);
        }
#pragma unroll
        for (int nt = 0; nt < 4; ++nt)
            acc[nt] = WMMA_F16(af, bf[nt], acc[nt]);

        // Finish staging, then one barrier per K-step.
#if HAVE_ASYNC
        if (more) __builtin_amdgcn_s_wait_asynccnt(0);
#else
        if (more) *(v8h*)(nxt + lofs) = wreg;
#endif
        __syncthreads();
    }
}

// ---------------------------------------------------------------------------
// QKV projection: qkv = x16 @ W_attn + b_attn, scattered head-major.
//   Q -> q16 [B,H,S,HD]   (pre-scaled by 1/sqrt(HD) = 0.125)
//   K -> k16 [B,H,S,HD]
//   V -> vt16 [B,H,HD,S]  (transposed, so PV B-fragments load contiguously)
// ---------------------------------------------------------------------------
__global__ __launch_bounds__(256) void qkv_kernel(const _Float16* __restrict__ x16,
                                                  const _Float16* __restrict__ watT,
                                                  const float* __restrict__ bias,
                                                  _Float16* __restrict__ q16,
                                                  _Float16* __restrict__ k16,
                                                  _Float16* __restrict__ vt16) {
    __shared__ _Float16 lds[2 * TILE_H];
    const int Nbase = blockIdx.x * 64;
    const int Mbase = blockIdx.y * 128;
    v8f acc[4] = {};
    gemm_mainloop(x16, watT, EE, Mbase, Nbase, lds, acc);

    const int lane = threadIdx.x & 31;
    const int wave = threadIdx.x >> 5;
    const int lm = lane & 15, lg = lane >> 4;
    const int sec   = Nbase / EE;          // 0=Q, 1=K, 2=V (uniform per block)
    const int ebase = Nbase - sec * EE;    // 64-aligned -> head uniform
    const int h     = ebase / HDD;

#pragma unroll
    for (int nt = 0; nt < 4; ++nt) {
        const int col = Nbase + nt * 16 + lm;
        const float bb = bias[col];
        const int d = nt * 16 + lm;        // ebase % 64 == 0, HD == 64
#pragma unroll
        for (int r = 0; r < 8; ++r) {
            const int row = Mbase + wave * 16 + r + 8 * lg;
            const int b = row >> 11;       // / S
            const int s = row & (SS - 1);
            const float v = acc[nt][r] + bb;
            const long bh = (long)(b * HH + h);
            if (sec == 0)
                q16[(bh * SS + s) * HDD + d] = (_Float16)(v * 0.125f);
            else if (sec == 1)
                k16[(bh * SS + s) * HDD + d] = (_Float16)v;
            else
                vt16[(bh * HDD + d) * SS + s] = (_Float16)v;
        }
    }
}

// ---------------------------------------------------------------------------
// Flash attention: causal + pad mask, online softmax, fp32 accumulators.
// Block = 128 q rows of one (b,h); wave w owns rows [qw, qw+16).
// 32-key blocks: 4 WMMA QK^T + 4 WMMA PV per block per wave.
// ---------------------------------------------------------------------------
__global__ __launch_bounds__(256) void attn_kernel(const _Float16* __restrict__ q16,
                                                   const _Float16* __restrict__ k16,
                                                   const _Float16* __restrict__ vt16,
                                                   const int* __restrict__ amask,
                                                   _Float16* __restrict__ attn16) {
    __shared__ _Float16 plds[8 * 16 * 32];     // wave-private P scratch
    const int bh   = blockIdx.y;               // 0..63
    const int b    = bh >> 4;                  // / H
    const int h    = bh & (HH - 1);
    const int wave = threadIdx.x >> 5;
    const int lane = threadIdx.x & 31;
    const int lm = lane & 15, lg = lane >> 4;
    const int qw = blockIdx.x * 128 + wave * 16;

    const _Float16* qp = q16 + (long)bh * SS * HDD;
    const _Float16* kp = k16 + (long)bh * SS * HDD;
    const _Float16* vp = vt16 + (long)bh * HDD * SS;
    const int* mp = amask + b * SS;
    _Float16* ps = plds + wave * 512;

    // Q fragments for both K-steps of the HD=64 contraction (scale pre-folded).
    v16h qa[2];
#pragma unroll
    for (int ks = 0; ks < 2; ++ks) {
        const _Float16* ap = qp + (long)(qw + lm) * HDD + ks * 32 + lg * 8;
        qa[ks] = load_frag_pair(ap, ap + 16);
    }

    v8f outacc[4] = {};
    float mrow[8], lrow[8];
#pragma unroll
    for (int r = 0; r < 8; ++r) { mrow[r] = -1e30f; lrow[r] = 0.0f; }

    const int kend = qw + 16;   // causal: keys <= max row of this wave
    for (int kb = 0; kb < kend; kb += 32) {
        // ----- all K fragments first, then 4 back-to-back WMMAs -----
        v16h kf[4];
#pragma unroll
        for (int nt = 0; nt < 2; ++nt)
#pragma unroll
            for (int ks = 0; ks < 2; ++ks) {
                const _Float16* bp =
                    kp + (long)(kb + nt * 16 + lm) * HDD + ks * 32 + lg * 8;
                kf[nt * 2 + ks] = load_frag_pair(bp, bp + 16);
            }
        v8f s[2] = {};
        s[0] = WMMA_F16(qa[0], kf[0], s[0]);
        s[0] = WMMA_F16(qa[1], kf[1], s[0]);
        s[1] = WMMA_F16(qa[0], kf[2], s[1]);
        s[1] = WMMA_F16(qa[1], kf[3], s[1]);

        // ----- causal + pad mask -----
#pragma unroll
        for (int nt = 0; nt < 2; ++nt) {
            const int col = kb + nt * 16 + lm;
            const bool pad_ok = (mp[col] != 0);
#pragma unroll
            for (int r = 0; r < 8; ++r) {
                const int row = qw + r + 8 * lg;
                if (col > row || !pad_ok) s[nt][r] = -1e30f;
            }
        }
        // ----- online softmax (row stats across 16-lane halves) -----
        v8f mx;
#pragma unroll
        for (int r = 0; r < 8; ++r) mx[r] = fmaxf(s[0][r], s[1][r]);
#pragma unroll
        for (int off = 1; off < 16; off <<= 1)
#pragma unroll
            for (int r = 0; r < 8; ++r)
                mx[r] = fmaxf(mx[r], __shfl_xor(mx[r], off, 32));

        v8f fac;
#pragma unroll
        for (int r = 0; r < 8; ++r) {
            const float mn = fmaxf(mrow[r], mx[r]);
            fac[r] = __expf(mrow[r] - mn);
            mrow[r] = mn;
        }
#pragma unroll
        for (int nt = 0; nt < 2; ++nt)
#pragma unroll
            for (int r = 0; r < 8; ++r)
                s[nt][r] = __expf(s[nt][r] - mrow[r]);

        v8f rs;
#pragma unroll
        for (int r = 0; r < 8; ++r) rs[r] = s[0][r] + s[1][r];
#pragma unroll
        for (int off = 1; off < 16; off <<= 1)
#pragma unroll
            for (int r = 0; r < 8; ++r)
                rs[r] += __shfl_xor(rs[r], off, 32);
#pragma unroll
        for (int r = 0; r < 8; ++r) lrow[r] = lrow[r] * fac[r] + rs[r];
#pragma unroll
        for (int nt = 0; nt < 4; ++nt)
#pragma unroll
            for (int r = 0; r < 8; ++r) outacc[nt][r] *= fac[r];

        // ----- re-layout P (C layout -> A fragment) via wave-private LDS -----
#pragma unroll
        for (int r = 0; r < 8; ++r) {
            ps[(r + 8 * lg) * 32 + lm]      = (_Float16)s[0][r];
            ps[(r + 8 * lg) * 32 + 16 + lm] = (_Float16)s[1][r];
        }
        const _Float16* pp = ps + lm * 32 + lg * 8;  // LDS in-order within wave
        const v16h pa = load_frag_pair(pp, pp + 16);

        // ----- all V fragments first, then 4 back-to-back WMMAs -----
        v16h vf[4];
#pragma unroll
        for (int nt = 0; nt < 4; ++nt) {
            const _Float16* bp = vp + (long)(nt * 16 + lm) * SS + kb + lg * 8;
            vf[nt] = load_frag_pair(bp, bp + 16);
        }
#pragma unroll
        for (int nt = 0; nt < 4; ++nt)
            outacc[nt] = WMMA_F16(pa, vf[nt], outacc[nt]);
    }

    // Epilogue: normalize and reassemble [B,S,E] (head-major columns), f16.
#pragma unroll
    for (int nt = 0; nt < 4; ++nt) {
#pragma unroll
        for (int r = 0; r < 8; ++r) {
            const int sq = qw + r + 8 * lg;
            const float o = outacc[nt][r] / lrow[r];
            attn16[((long)(b * SS + sq)) * EE + h * HDD + nt * 16 + lm] = (_Float16)o;
        }
    }
}

// ---------------------------------------------------------------------------
// Output projection: out = attn16 @ W_proj + b_proj  (fp32 result)
// ---------------------------------------------------------------------------
__global__ __launch_bounds__(256) void proj_kernel(const _Float16* __restrict__ a16,
                                                   const _Float16* __restrict__ wprT,
                                                   const float* __restrict__ bias,
                                                   float* __restrict__ out) {
    __shared__ _Float16 lds[2 * TILE_H];
    const int Nbase = blockIdx.x * 64;
    const int Mbase = blockIdx.y * 128;
    v8f acc[4] = {};
    gemm_mainloop(a16, wprT, EE, Mbase, Nbase, lds, acc);

    const int lane = threadIdx.x & 31;
    const int wave = threadIdx.x >> 5;
    const int lm = lane & 15, lg = lane >> 4;
#pragma unroll
    for (int nt = 0; nt < 4; ++nt) {
        const int col = Nbase + nt * 16 + lm;
        const float bb = bias[col];
#pragma unroll
        for (int r = 0; r < 8; ++r) {
            const int row = Mbase + wave * 16 + r + 8 * lg;
            out[(long)row * EE + col] = acc[nt][r] + bb;
        }
    }
}

// ---------------------------------------------------------------------------
// Host launcher
// ---------------------------------------------------------------------------
extern "C" void kernel_launch(void* const* d_in, const int* in_sizes, int n_in,
                              void* d_out, int out_size, void* d_ws, size_t ws_size,
                              hipStream_t stream) {
    const float* x      = (const float*)d_in[0];
    const int*   amask  = (const int*)d_in[1];
    const float* W_attn = (const float*)d_in[2];
    const float* b_attn = (const float*)d_in[3];
    const float* W_proj = (const float*)d_in[4];
    const float* b_proj = (const float*)d_in[5];
    float* out = (float*)d_out;

    // Workspace carve-up (all sizes are multiples of 256 bytes).
    char* ws = (char*)d_ws;
    const size_t nME   = (size_t)MM * EE;      // 8388608
    const size_t nWatt = (size_t)EE * NE3;     // 3145728
    const size_t nWpro = (size_t)EE * EE;      // 1048576
    _Float16* x16    = (_Float16*)ws;                  ws += nME * 2;
    _Float16* watT   = (_Float16*)ws;                  ws += nWatt * 2;   // [3E][E]
    _Float16* wprT   = (_Float16*)ws;                  ws += nWpro * 2;   // [E][E]
    _Float16* q16    = (_Float16*)ws;                  ws += nME * 2;
    _Float16* k16    = (_Float16*)ws;                  ws += nME * 2;
    _Float16* vt16   = (_Float16*)ws;                  ws += nME * 2;
    _Float16* attn16 = (_Float16*)ws;                  ws += nME * 2;

    // 1) fp32 -> f16 conversions (x straight; weights transposed to [N][K])
    {
        const int n4 = (int)(nME / 4);
        cvt_kernel<<<(n4 + 255) / 256, 256, 0, stream>>>((const float4*)x, (v4h*)x16, n4);
        cvtT_kernel<<<dim3(NE3 / 64, EE / 64), 256, 0, stream>>>(W_attn, watT, EE, NE3);
        cvtT_kernel<<<dim3(EE / 64, EE / 64), 256, 0, stream>>>(W_proj, wprT, EE, EE);
    }
    // 2) QKV projection + head scatter
    qkv_kernel<<<dim3(NE3 / 64, MM / 128), 256, 0, stream>>>(x16, watT, b_attn, q16, k16, vt16);
    // 3) Flash attention
    attn_kernel<<<dim3(SS / 128, BB * HH), 256, 0, stream>>>(q16, k16, vt16, amask, attn16);
    // 4) Output projection
    proj_kernel<<<dim3(EE / 64, MM / 128), 256, 0, stream>>>(attn16, wprT, b_proj, out);
}
